// GATModel_90460601189315
// MI455X (gfx1250) — matrix operand order repaired and checked by reading, not compiled, not run on hardware
//
#include <hip/hip_runtime.h>
#include <hip/hip_bf16.h>
#include <math.h>

typedef __attribute__((ext_vector_type(16))) _Float16 v16h;
typedef __attribute__((ext_vector_type(8)))  _Float16 v8h;
typedef __attribute__((ext_vector_type(8)))  float    v8f;

union Frag16 { v16h v; v8h h[2]; };

#define NEG_SLOPE 0.2f
#define EPSV 1e-9f

// ---------------- small utility kernels ----------------

__global__ void fill_f32(float* __restrict__ p, float v, int count) {
  int i = blockIdx.x * blockDim.x + threadIdx.x;
  if (i < count) p[i] = v;
}

__global__ void cvt_f16(const float* __restrict__ src, _Float16* __restrict__ dst, int count) {
  int i = blockIdx.x * blockDim.x + threadIdx.x;
  if (i < count) dst[i] = (_Float16)src[i];
}

// Build transposed f16 weights:
//  WT1[c][d] (64 x 128) from W1[h=4][d=128][f=16], c = h*16+f
//  WT2[c][d] (16 x 64)  from W2[1][d=64][f=16],   c = f
__global__ void prep_w(const float* __restrict__ W1, const float* __restrict__ W2,
                       _Float16* __restrict__ WT1, _Float16* __restrict__ WT2) {
  int i = blockIdx.x * blockDim.x + threadIdx.x;
  if (i < 64 * 128) {
    int c = i >> 7, d = i & 127;
    WT1[i] = (_Float16)W1[(c >> 4) * (128 * 16) + d * 16 + (c & 15)];
  } else if (i < 64 * 128 + 16 * 64) {
    int j = i - 64 * 128;
    int c = j >> 6, d = j & 63;
    WT2[j] = (_Float16)W2[d * 16 + c];
  }
}

// ---------------- WMMA GEMM: C[M,N] = A[M,K] * BT[N,K]^T ----------------
// A, BT in f16; C in f32. One wave computes a 16-row stripe across all N.
template <int NT, int KS>
__global__ void __launch_bounds__(256)
gemm_wmma(const _Float16* __restrict__ A, const _Float16* __restrict__ BT,
          float* __restrict__ C, int M) {
  constexpr int K = KS * 32;
  constexpr int N = NT * 16;
  int lane = threadIdx.x & 31;
  int wave = threadIdx.x >> 5;
  int row0 = (blockIdx.x * (int)(blockDim.x >> 5) + wave) * 16;
  if (row0 >= M) return;                       // wave-uniform: EXEC stays all-ones
  int g  = lane >> 4;                          // half-wave group
  int mr = lane & 15;
  int ar = row0 + mr; if (ar >= M) ar = M - 1; // clamp (stores are guarded)
  const _Float16* arow = A + (size_t)ar * K;

  v8f acc[NT];
#pragma unroll
  for (int t = 0; t < NT; ++t)
#pragma unroll
    for (int i = 0; i < 8; ++i) acc[t][i] = 0.0f;

#pragma unroll
  for (int ks = 0; ks < KS; ++ks) {
    const int k0 = ks * 32;
    Frag16 a;
    a.h[0] = *(const v8h*)(arow + k0 + 8 * g);        // K = k0+8g .. +7
    a.h[1] = *(const v8h*)(arow + k0 + 16 + 8 * g);   // K = k0+16+8g .. +7
#pragma unroll
    for (int t = 0; t < NT; ++t) {
      Frag16 b;
      const _Float16* bp = BT + (size_t)(t * 16 + mr) * K + k0 + 8 * g;
      b.h[0] = *(const v8h*)(bp);                     // K = k0+8g .. +7
      b.h[1] = *(const v8h*)(bp + 16);                // K = k0+16+8g .. +7
      acc[t] = __builtin_amdgcn_wmma_f32_16x16x32_f16(
          false, a.v, false, b.v, (short)0, acc[t], false, false);
    }
  }

#pragma unroll
  for (int t = 0; t < NT; ++t)
#pragma unroll
    for (int i = 0; i < 8; ++i) {
      int r = row0 + i + 8 * g;
      if (r < M) C[(size_t)r * N + t * 16 + mr] = acc[t][i];
    }
}

// ---------------- attention kernels ----------------

__device__ __forceinline__ float atomicMaxFloat(float* addr, float val) {
  if (val >= 0.0f)
    return __int_as_float(atomicMax((int*)addr, __float_as_int(val)));
  else
    return __uint_as_float(atomicMin((unsigned int*)addr, __float_as_uint(val)));
}

// s_src[n,H], s_dst[n,H] from h[n,H*F], a_src/a_dst [H,F]
__global__ void scores_k(const float* __restrict__ h, const float* __restrict__ asrc,
                         const float* __restrict__ adst, float* __restrict__ ss,
                         float* __restrict__ sd, int count, int H, int F) {
  int t = blockIdx.x * blockDim.x + threadIdx.x;
  if (t >= count) return;
  int nd = t / H, hh = t - nd * H;
  const float* hp = h + (size_t)nd * H * F + hh * F;
  float s1 = 0.f, s2 = 0.f;
  for (int f = 0; f < F; ++f) {
    float v = hp[f];
    s1 += v * asrc[hh * F + f];
    s2 += v * adst[hh * F + f];
  }
  ss[t] = s1; sd[t] = s2;
}

// e = leaky_relu(ss[s]+sd[r]); stash e; segment-max into m[r]
__global__ void edge_max_k(const int* __restrict__ snd, const int* __restrict__ rcv,
                           const float* __restrict__ ss, const float* __restrict__ sd,
                           float* __restrict__ eb, float* __restrict__ m,
                           int count, int H) {
  int t = blockIdx.x * blockDim.x + threadIdx.x;
  if (t >= count) return;
  int e = t / H, h = t - e * H;
  int s = snd[e], r = rcv[e];
  float v = ss[s * H + h] + sd[r * H + h];
  v = v > 0.f ? v : NEG_SLOPE * v;
  eb[t] = v;
  atomicMaxFloat(&m[r * H + h], v);
}

// alpha = exp(e - m[r]); stash alpha (in place); segment-sum into denom[r]
__global__ void edge_alpha_k(const int* __restrict__ rcv, float* __restrict__ eb,
                             const float* __restrict__ m, float* __restrict__ dn,
                             int count, int H) {
  int t = blockIdx.x * blockDim.x + threadIdx.x;
  if (t >= count) return;
  int e = t / H, h = t - e * H;
  int r = rcv[e];
  float a = __expf(eb[t] - m[r * H + h]);
  eb[t] = a;
  atomicAdd(&dn[r * H + h], a);
}

// agg[r, c] += alpha[e, c/F] * h[s, c]   (one thread per (edge, column))
__global__ void edge_agg_k(const int* __restrict__ snd, const int* __restrict__ rcv,
                           const float* __restrict__ alpha, const float* __restrict__ h,
                           float* __restrict__ agg, int count, int H, int F) {
  int t = blockIdx.x * blockDim.x + threadIdx.x;
  if (t >= count) return;
  int C = H * F;
  int e = t / C, c = t - e * C;
  int hh = c / F;
  float a = alpha[e * H + hh];
  atomicAdd(&agg[(size_t)rcv[e] * C + c], a * h[(size_t)snd[e] * C + c]);
}

// out1_f16 = elu(agg / (denom + eps))   (layer-1 epilogue, feeds gemm2)
__global__ void node_elu_k(const float* __restrict__ agg, const float* __restrict__ dn,
                           _Float16* __restrict__ oh, int count) {
  int t = blockIdx.x * blockDim.x + threadIdx.x;
  if (t >= count) return;
  int nd = t >> 6, c = t & 63;
  float v = agg[t] / (dn[nd * 4 + (c >> 4)] + EPSV);
  v = v > 0.f ? v : expm1f(v);
  oh[t] = (_Float16)v;
}

// final: out /= (denom2 + eps)  (in place in d_out)
__global__ void node_out_k(float* __restrict__ out, const float* __restrict__ dn, int count) {
  int t = blockIdx.x * blockDim.x + threadIdx.x;
  if (t >= count) return;
  out[t] = out[t] / (dn[t >> 4] + EPSV);
}

// ---------------- launch ----------------

extern "C" void kernel_launch(void* const* d_in, const int* in_sizes, int n_in,
                              void* d_out, int out_size, void* d_ws, size_t ws_size,
                              hipStream_t stream) {
  const float* x   = (const float*)d_in[0];
  const int*   snd = (const int*)d_in[1];
  const int*   rcv = (const int*)d_in[2];
  const float* W1  = (const float*)d_in[3];
  const float* a1s = (const float*)d_in[4];
  const float* a1d = (const float*)d_in[5];
  const float* W2  = (const float*)d_in[6];
  const float* a2s = (const float*)d_in[7];
  const float* a2d = (const float*)d_in[8];
  float* out = (float*)d_out;

  const int n = in_sizes[0] / 128;
  const int E = in_sizes[1];

  char* w = (char*)d_ws;
  size_t off = 0;
  auto carve = [&](size_t bytes) -> char* {
    char* p = w + off;
    off = (off + bytes + 255) & ~(size_t)255;
    return p;
  };
  _Float16* WT1  = (_Float16*)carve(64 * 128 * 2);
  _Float16* WT2  = (_Float16*)carve(16 * 64 * 2);
  _Float16* xh   = (_Float16*)carve((size_t)n * 128 * 2);
  float*    h1   = (float*)carve((size_t)n * 64 * 4);
  float*    ss1  = (float*)carve((size_t)n * 4 * 4);
  float*    sd1  = (float*)carve((size_t)n * 4 * 4);
  float*    m1   = (float*)carve((size_t)n * 4 * 4);
  float*    dn1  = (float*)carve((size_t)n * 4 * 4);
  float*    eb1  = (float*)carve((size_t)E * 4 * 4);
  float*    agg1 = (float*)carve((size_t)n * 64 * 4);
  // phase-safe aliases for layer 2
  _Float16* o1h = xh;           // x_f16 dead after gemm1
  float*    h2  = h1;           // h1 dead after layer-1 aggregation
  float*    ss2 = ss1; float* sd2 = sd1;
  float*    m2  = m1;  float* dn2 = dn1;
  float*    eb2 = eb1;

  const int T = 256;
  auto G = [](long long c) { return dim3((unsigned)((c + 255) / 256)); };

  // ---- layer 1 ----
  fill_f32<<<G((long long)n * 4), T, 0, stream>>>(m1, -INFINITY, n * 4);
  fill_f32<<<G((long long)n * 4), T, 0, stream>>>(dn1, 0.f, n * 4);
  fill_f32<<<G((long long)n * 64), T, 0, stream>>>(agg1, 0.f, n * 64);
  prep_w<<<G(64 * 128 + 16 * 64), T, 0, stream>>>(W1, W2, WT1, WT2);
  cvt_f16<<<G((long long)n * 128), T, 0, stream>>>(x, xh, n * 128);

  gemm_wmma<4, 4><<<dim3((n + 127) / 128), T, 0, stream>>>(xh, WT1, h1, n);

  scores_k<<<G((long long)n * 4), T, 0, stream>>>(h1, a1s, a1d, ss1, sd1, n * 4, 4, 16);
  edge_max_k<<<G((long long)E * 4), T, 0, stream>>>(snd, rcv, ss1, sd1, eb1, m1, E * 4, 4);
  edge_alpha_k<<<G((long long)E * 4), T, 0, stream>>>(rcv, eb1, m1, dn1, E * 4, 4);
  edge_agg_k<<<G((long long)E * 64), T, 0, stream>>>(snd, rcv, eb1, h1, agg1, E * 64, 4, 16);
  node_elu_k<<<G((long long)n * 64), T, 0, stream>>>(agg1, dn1, o1h, n * 64);

  // ---- layer 2 (init after layer-1 readers of aliased buffers are done) ----
  fill_f32<<<G((long long)n), T, 0, stream>>>(m2, -INFINITY, n);
  fill_f32<<<G((long long)n), T, 0, stream>>>(dn2, 0.f, n);
  fill_f32<<<G((long long)n * 16), T, 0, stream>>>(out, 0.f, n * 16);

  gemm_wmma<1, 2><<<dim3((n + 127) / 128), T, 0, stream>>>(o1h, WT2, h2, n);

  scores_k<<<G((long long)n), T, 0, stream>>>(h2, a2s, a2d, ss2, sd2, n, 1, 16);
  edge_max_k<<<G((long long)E), T, 0, stream>>>(snd, rcv, ss2, sd2, eb2, m2, E, 1);
  edge_alpha_k<<<G((long long)E), T, 0, stream>>>(rcv, eb2, m2, dn2, E, 1);
  edge_agg_k<<<G((long long)E * 16), T, 0, stream>>>(snd, rcv, eb2, h2, out, E * 16, 1, 16);
  node_out_k<<<G((long long)n * 16), T, 0, stream>>>(out, dn2, n * 16);
}
